// DCN_out_22136261444435
// MI455X (gfx1250) — compile-verified, matrix operand order checked by reading, not run on hardware
//
#include <hip/hip_runtime.h>
#include <hip/hip_bf16.h>
#include <math.h>

// ---------------------------------------------------------------------------
// Fused DCN forward for MI455X (gfx1250, wave32, WMMA bf16)
//   B=8, C=64, H=W=256, k=3  (shapes fixed by the reference setup)
//
// Per workgroup (1 batch b, 1 output row h, 32-wide w tile), 256 thr = 8 waves:
//   Ph0: stage w_dcn -> sWb (bf16 [64][192]), w_off -> sWoffb (bf16 [16][192],
//        rows 9..15 zero), zero LDS stats; prefetch h+/-2 rows (global_prefetch)
//   Ph1: stage x rows h-1..h+1 as bf16 B-operand sB[p][c*3+i] (zero-padded H)
//   Ph2: OFFSET CONV AS WMMA GEMM: om(16x32) = Woffb(16x192) x sB(192x32),
//        waves 0..1, 6 k-steps of v_wmma_f32_16x16x32_bf16; +b_off, sigmoid
//   Ph3: bilinear gather from global x (entire x = 134MB < 192MB L2),
//        coords hoisted per (kk,p); overwrite sB with sampled*mask (bf16)
//   Ph4: MAIN GEMM: out(64x32) = sWb(64x192) x sB(192x32), 8 waves x 1 tile
//   Epilogue: +bias, store pre-BN, per-channel sum/sumsq -> LDS -> 32-slot
//             global bins (atomic contention 512/addr instead of 16K/addr)
// Then: finalize BN scale/shift (1 tiny block); apply BN+ReLU in place.
// ---------------------------------------------------------------------------

typedef __attribute__((ext_vector_type(16))) __bf16 v16bf;
typedef __attribute__((ext_vector_type(8)))  float  v8f;

#define Bsz 8
#define Cch 64
#define Hh  256
#define Ww  256
#define KK  3
#define KDIM 192            // C*k
#define NT  32              // w-tile width (GEMM N per workgroup)
#define SROW 200            // padded bf16 row stride (16B aligned, conflict pad)
#define NSLOT 32            // stat accumulation slots
// ws layout (floats): [0 .. NSLOT*128)  : per-slot {sum[64], sumsq[64]}
//                     [4096..4159]      : scale[c]
//                     [4160..4223]      : shift[c]

union FragU { v16bf v; uint4 q[2]; };

// 16x16 D-tile = A(16x192) x B(192x16), K swept in 6 WMMA steps.
// Per-lane K layout (ISA 7.12.2): lanes 0-15 carry K {0..7,16..23}+32kk,
// lanes 16-31 carry K {8..15,24..31}+32kk.  A row / B col = lane&15.
__device__ __forceinline__ v8f gemm192(const __bf16 (*A)[SROW], int row,
                                       const __bf16 (*B)[SROW], int col,
                                       int khalf) {
    v8f acc = {};
    #pragma unroll
    for (int kk = 0; kk < KDIM / 32; ++kk) {
        int kb = kk * 32 + khalf;
        FragU a, bb;
        a.q[0]  = *reinterpret_cast<const uint4*>(&A[row][kb]);
        a.q[1]  = *reinterpret_cast<const uint4*>(&A[row][kb + 16]);
        bb.q[0] = *reinterpret_cast<const uint4*>(&B[col][kb]);
        bb.q[1] = *reinterpret_cast<const uint4*>(&B[col][kb + 16]);
        acc = __builtin_amdgcn_wmma_f32_16x16x32_bf16(
                  false, a.v, false, bb.v, (short)0, acc, false, false);
    }
    return acc;
}

__global__ __launch_bounds__(256)
void dcn_zero_stats(float* ws) {
    int t = threadIdx.x;
    #pragma unroll
    for (int j = 0; j < NSLOT * 128 / 256; ++j)
        ws[t + j * 256] = 0.0f;
}

__global__ __launch_bounds__(256)
void dcn_fused(const float* __restrict__ x,
               const float* __restrict__ w_off,
               const float* __restrict__ b_off,
               const float* __restrict__ w_dcn,
               const float* __restrict__ b_dcn,
               float* __restrict__ out,
               float* __restrict__ ws)
{
    __shared__ __align__(16) __bf16 sWb[Cch][SROW];    // 25600 B  main A
    __shared__ __align__(16) __bf16 sWoffb[16][SROW];  //  6400 B  offset A (M padded)
    __shared__ __align__(16) __bf16 sB[NT][SROW];      // 12800 B  X, then sampled
    __shared__ float sOm[9][NT];                       //  1152 B
    __shared__ float sMask[KK][NT];                    //   384 B
    __shared__ float sSum[Cch], sSq[Cch];              //   512 B

    const int t   = threadIdx.x;
    const int bid = blockIdx.x;                  // 16384 blocks
    const int wt  = bid & 7;
    const int h   = (bid >> 3) & 255;
    const int b   = bid >> 11;
    const int w0  = wt * NT;
    const int lane = t & 31;
    const int wv   = t >> 5;

    // ---- Ph0: weights -> LDS (bf16), stats init, L2 warm-up ------------
    #pragma unroll
    for (int j = 0; j < 48; ++j) {               // 64*192 w_dcn
        int e = t + j * 256;
        sWb[e / KDIM][e % KDIM] = (__bf16)w_dcn[e];
    }
    #pragma unroll
    for (int j = 0; j < 12; ++j) {               // 16*192 w_off (padded M)
        int e = t + j * 256;
        int o = e / KDIM, kid = e % KDIM;
        sWoffb[o][kid] = (o < 9) ? (__bf16)w_off[e] : (__bf16)0.0f;
    }
    if (t < Cch) { sSum[t] = 0.0f; sSq[t] = 0.0f; }
    if (t < 128) {                               // prefetch rows h-2, h+2
        int c = t & 63;
        int rr = (t >> 6) ? h + 2 : h - 2;
        if (rr >= 0 && rr < Hh)
            __builtin_prefetch(&x[(((size_t)b * Cch + c) * Hh + rr) * Ww + w0], 0, 2);
    }

    // ---- Ph1: x rows h-1..h+1 -> bf16 B-operand sB[p][c*3+i] -----------
    #pragma unroll
    for (int j = 0; j < 24; ++j) {               // 64*3*32 elements
        int e = t + j * 256;
        int c = e / (3 * NT);
        int rem = e % (3 * NT);
        int i = rem / NT, p = rem % NT;
        int r = h - 1 + i;
        float v = 0.0f;
        if (r >= 0 && r < Hh)
            v = x[(((size_t)b * Cch + c) * Hh + r) * Ww + w0 + p];
        sB[p][c * 3 + i] = (__bf16)v;
    }
    __syncthreads();

    // ---- Ph2: offset conv as WMMA (waves 0..1) -------------------------
    if (wv < 2) {                                // wave-uniform: EXEC stays full
        const int col = wv * 16 + (lane & 15);
        const int khalf = (lane >> 4) * 8;
        v8f oacc = gemm192(sWoffb, lane & 15, sB, col, khalf);
        #pragma unroll
        for (int r = 0; r < 8; ++r) {
            int o = r + ((lane >> 4) << 3);
            if (o < 9) {
                float v = oacc[r] + b_off[o];
                sOm[o][col] = v;
                if (o >= 6) sMask[o - 6][col] = 1.0f / (1.0f + __expf(-v));
            }
        }
    }
    __syncthreads();                             // sB (X) dead after this

    // ---- Ph3: bilinear gather -> sampled*mask (bf16) into sB -----------
    {
        int pr = t & 127;                        // 96 (kk,p) pairs, 2 c-halves
        int kk = pr >> 5, p = pr & 31;
        int c0 = (t >> 7) * 32;
        if (kk < KK) {
            float dy = sOm[kk][p], dx = sOm[3 + kk][p], m = sMask[kk][p];
            float yp  = (float)(h + kk) + dy;    // padded-H coordinates
            float xcf = (float)(w0 + p) + dx;
            float y0f = floorf(yp), x0f = floorf(xcf);
            int   y0 = (int)y0f,   x0 = (int)x0f;
            float wy1 = yp - y0f, wy0 = 1.0f - wy1;
            float wx1 = xcf - x0f, wx0 = 1.0f - wx1;
            int ro0 = y0 - 1, ro1 = y0;          // padded row -> original row
            bool vy0 = (unsigned)ro0 < (unsigned)Hh, vy1 = (unsigned)ro1 < (unsigned)Hh;
            bool vx0 = (unsigned)x0 < (unsigned)Ww, vx1 = (unsigned)(x0 + 1) < (unsigned)Ww;
            int ry0 = min(max(ro0, 0), Hh - 1), ry1 = min(max(ro1, 0), Hh - 1);
            int rx0 = min(max(x0, 0), Ww - 1),  rx1 = min(max(x0 + 1, 0), Ww - 1);
            float w00 = wy0 * wx0 * ((vy0 && vx0) ? 1.0f : 0.0f);
            float w01 = wy0 * wx1 * ((vy0 && vx1) ? 1.0f : 0.0f);
            float w10 = wy1 * wx0 * ((vy1 && vx0) ? 1.0f : 0.0f);
            float w11 = wy1 * wx1 * ((vy1 && vx1) ? 1.0f : 0.0f);
            int o00 = ry0 * Ww + rx0, o01 = ry0 * Ww + rx1;
            int o10 = ry1 * Ww + rx0, o11 = ry1 * Ww + rx1;
            const float* __restrict__ bx = x + ((size_t)b * Cch + c0) * (Hh * Ww);
            #pragma unroll 4
            for (int cc = 0; cc < 32; ++cc) {
                float v = bx[o00] * w00 + bx[o01] * w01
                        + bx[o10] * w10 + bx[o11] * w11;
                sB[p][(c0 + cc) * 3 + kk] = (__bf16)(v * m);
                bx += Hh * Ww;
            }
        }
    }
    __syncthreads();

    // ---- Ph4: main WMMA GEMM  out(64x32) = sWb x sB --------------------
    const int mtile = wv >> 1;                   // 0..3
    const int ntile = wv & 1;                    // 0..1
    const int khalf = (lane >> 4) * 8;
    v8f acc = gemm192(sWb, mtile * 16 + (lane & 15), sB, ntile * 16 + (lane & 15), khalf);

    // ---- Epilogue: +bias, store pre-BN, accumulate BN stats ------------
    const int pcol = ntile * 16 + (lane & 15);
    #pragma unroll
    for (int r = 0; r < 8; ++r) {
        int o = mtile * 16 + r + ((lane >> 4) << 3);
        float v = acc[r] + b_dcn[o];
        out[(((size_t)b * Cch + o) * Hh + h) * Ww + w0 + pcol] = v;
        atomicAdd(&sSum[o], v);
        atomicAdd(&sSq[o], v * v);
    }
    __syncthreads();
    if (t < Cch) {
        int slot = bid & (NSLOT - 1);
        atomicAdd(&ws[slot * 128 + t],      sSum[t]);
        atomicAdd(&ws[slot * 128 + 64 + t], sSq[t]);
    }
}

__global__ __launch_bounds__(64)
void dcn_bn_finalize(float* ws,
                     const float* __restrict__ gamma,
                     const float* __restrict__ beta)
{
    int c = threadIdx.x;
    if (c >= Cch) return;
    float sum = 0.0f, sq = 0.0f;
    #pragma unroll
    for (int s = 0; s < NSLOT; ++s) {
        sum += ws[s * 128 + c];
        sq  += ws[s * 128 + 64 + c];
    }
    const float N = (float)Bsz * Hh * Ww;        // 524288
    float mean = sum / N;
    float var  = sq / N - mean * mean;
    float rs   = rsqrtf(var + 1e-5f);
    float scale = gamma[c] * rs;
    ws[4096 + c] = scale;
    ws[4160 + c] = beta[c] - mean * scale;
}

__global__ __launch_bounds__(256)
void dcn_bn_relu(float* __restrict__ out, const float* __restrict__ ws) {
    int i4 = blockIdx.x * 256 + threadIdx.x;     // float4 index
    const int total4 = (Bsz * Cch * Hh * Ww) / 4;
    if (i4 >= total4) return;
    int c = (i4 >> 14) & 63;                     // H*W/4 = 16384 float4 / channel
    float scale = ws[4096 + c], shift = ws[4160 + c];
    float4* p = reinterpret_cast<float4*>(out) + i4;
    float4 v = *p;
    v.x = fmaxf(0.0f, v.x * scale + shift);
    v.y = fmaxf(0.0f, v.y * scale + shift);
    v.z = fmaxf(0.0f, v.z * scale + shift);
    v.w = fmaxf(0.0f, v.w * scale + shift);
    *p = v;
}

extern "C" void kernel_launch(void* const* d_in, const int* in_sizes, int n_in,
                              void* d_out, int out_size, void* d_ws, size_t ws_size,
                              hipStream_t stream) {
    (void)in_sizes; (void)n_in; (void)out_size; (void)ws_size;
    const float* x     = (const float*)d_in[0];
    const float* w_off = (const float*)d_in[1];
    const float* b_off = (const float*)d_in[2];
    const float* w_dcn = (const float*)d_in[3];
    const float* b_dcn = (const float*)d_in[4];
    const float* gamma = (const float*)d_in[5];
    const float* beta  = (const float*)d_in[6];
    float* out = (float*)d_out;
    float* ws  = (float*)d_ws;

    dcn_zero_stats<<<1, 256, 0, stream>>>(ws);

    const int nblocks = Bsz * Hh * (Ww / NT);    // 16384
    dcn_fused<<<nblocks, 256, 0, stream>>>(x, w_off, b_off, w_dcn, b_dcn, out, ws);

    dcn_bn_finalize<<<1, 64, 0, stream>>>(ws, gamma, beta);

    const int total4 = (Bsz * Cch * Hh * Ww) / 4;
    dcn_bn_relu<<<(total4 + 255) / 256, 256, 0, stream>>>(out, ws);
}
